// SoftPool_1d_30219389895196
// MI455X (gfx1250) — matrix operand order repaired and checked by reading, not compiled, not run on hardware
//
#include <hip/hip_runtime.h>

typedef __attribute__((ext_vector_type(16))) _Float16     v16h;
typedef __attribute__((ext_vector_type(8)))  float        v8f;
typedef __attribute__((ext_vector_type(2)))  _Float16     h2;
typedef __attribute__((ext_vector_type(4)))  float        f4;
typedef __attribute__((ext_vector_type(4)))  unsigned int u32x4;
typedef __attribute__((ext_vector_type(8)))  int          i32x8;
typedef __attribute__((ext_vector_type(4)))  int          i32x4;

#define D        256   // feature dim (K of GEMM)
#define HID      512   // hidden dim (N of GEMM)
#define ROWS     64    // rows of x per block = 16 windows
#define KC       32    // K chunk per WMMA pass
#define NKC      (D / KC)

// dynamic LDS layout (bytes):
//   float    xs[64][256]        65536  f32 x tile (epilogue reuse)
//   _Float16 xh[64][256]        32768  f16 x tile (WMMA A)
//   float    wstage[2][32][512] 131072 TDM staging for We chunks (double buffered)
//   _Float16 wt[2][512][32]     65536  transposed f16 We chunks (WMMA B)
//   float    logits[64]         256
#define SM_XS   0
#define SM_XH   (SM_XS + ROWS * D * 4)
#define SM_WST  (SM_XH + ROWS * D * 2)
#define SM_WT   (SM_WST + 2 * KC * HID * 4)
#define SM_LG   (SM_WT + 2 * HID * KC * 2)
#define SM_SIZE (SM_LG + ROWS * 4)

#if defined(__HIP_DEVICE_COMPILE__) && defined(__gfx1250__) && \
    __has_builtin(__builtin_amdgcn_tensor_load_to_lds)
#define HAVE_TDM 1
#endif

#ifdef HAVE_TDM
// Issue one TDM 2D tile load: contiguous (dim1 rows x dim0 elems) of f32,
// tile == tensor, stride0 in elements.  D# per ISA ch.8 (groups 0/1; 2/3 zero).
// clang-23 toolchain: 6-arg builtin (g0, g1, g2, g3, extra_g, cpol).
__device__ __forceinline__ void tdm_load_2d_f32(unsigned lds_byte_addr,
                                                const float* gptr,
                                                unsigned dim0, unsigned dim1,
                                                unsigned stride0) {
    unsigned long long ga = (unsigned long long)(uintptr_t)gptr;
    u32x4 g0;
    g0.x = 1u;                                            // count=1 valid descriptor
    g0.y = lds_byte_addr;                                 // lds_addr[63:32]
    g0.z = (unsigned)(ga & 0xFFFFFFFFull);                // global_addr[95:64]
    g0.w = (unsigned)((ga >> 32) & 0x1FFFFFFull)          // global_addr[120:96]
         | 0x80000000u;                                   // type=2 ("image") [127:126]
    i32x8 g1;
    g1[0] = (int)(2u << 16);                              // data_size=4B; mask/pad=0
    g1[1] = (int)((dim0 & 0xFFFFu) << 16);                // tensor_dim0 lo16 @ [63:48]
    g1[2] = (int)(((dim0 >> 16) & 0xFFFFu) | ((dim1 & 0xFFFFu) << 16)); // dim0 hi | dim1 lo
    g1[3] = (int)(((dim1 >> 16) & 0xFFFFu) | ((dim0 & 0xFFFFu) << 16)); // dim1 hi | tile_dim0
    g1[4] = (int)(dim1 & 0xFFFFu);                        // tile_dim1 | tile_dim2=0
    g1[5] = (int)stride0;                                 // tensor_dim0_stride lo32
    g1[6] = 0;                                            // stride0 hi | stride1 lo
    g1[7] = 0;
    i32x4 z4; z4[0] = 0; z4[1] = 0; z4[2] = 0; z4[3] = 0;
    i32x8 z8;
    #pragma unroll
    for (int i = 0; i < 8; ++i) z8[i] = 0;
    __builtin_amdgcn_tensor_load_to_lds(g0, g1, z4, z4, z8, 0);
}
#endif

__device__ __forceinline__ float fast_tanh(float v) {
    // tanh(v) = 1 - 2/(e^{2v}+1); exact limits at +/-inf via v_exp_f32
    float e = __expf(2.0f * v);
    return 1.0f - 2.0f / (e + 1.0f);
}

__global__ __launch_bounds__(256)
void softpool1d_wmma_kernel(const float* __restrict__ x,
                            const float* __restrict__ We,
                            const float* __restrict__ be,
                            const float* __restrict__ Wr,
                            const float* __restrict__ br,
                            float* __restrict__ out)
{
    extern __shared__ char smem[];
    float*    xs     = (float*)(smem + SM_XS);
    _Float16* xh     = (_Float16*)(smem + SM_XH);
    float*    wstage = (float*)(smem + SM_WST);     // [2][KC*HID]
    _Float16* wt     = (_Float16*)(smem + SM_WT);   // [2][HID*KC]
    float*    logits = (float*)(smem + SM_LG);

    const int tid  = threadIdx.x;
    const int wave = tid >> 5;
    const int lane = tid & 31;
    const int half = lane >> 4;
    const int l16  = lane & 15;
    const int blk  = blockIdx.x;

    if (tid < ROWS) logits[tid] = 0.0f;

    const float* xg = x + (size_t)blk * (ROWS * D);

    // ---------------- prologue: stage x tile + We chunk 0 ----------------
#ifdef HAVE_TDM
    if (wave == 0) {
        tdm_load_2d_f32((unsigned)(uintptr_t)xs, xg, D, ROWS, D);
        tdm_load_2d_f32((unsigned)(uintptr_t)wstage, We, HID, KC, HID);
        __builtin_amdgcn_s_wait_tensorcnt(0);
    }
    __syncthreads();
    // convert f32 x tile -> f16 A-operand copy
    for (int i = tid; i < (ROWS * D) / 4; i += 256) {
        f4 v = ((const f4*)xs)[i];
        h2 p0; p0.x = (_Float16)v.x; p0.y = (_Float16)v.y;
        h2 p1; p1.x = (_Float16)v.z; p1.y = (_Float16)v.w;
        ((h2*)xh)[2 * i]     = p0;
        ((h2*)xh)[2 * i + 1] = p1;
    }
    // convert chunk 0 staging -> transposed f16 B-operand
    for (int i = tid; i < KC * HID; i += 256) {
        int k = i >> 9;
        int n = i & (HID - 1);
        wt[n * KC + k] = (_Float16)wstage[i];
    }
#else
    for (int i = tid; i < (ROWS * D) / 4; i += 256) {
        f4 v = ((const f4*)xg)[i];
        ((f4*)xs)[i] = v;
        h2 p0; p0.x = (_Float16)v.x; p0.y = (_Float16)v.y;
        h2 p1; p1.x = (_Float16)v.z; p1.y = (_Float16)v.w;
        ((h2*)xh)[2 * i]     = p0;
        ((h2*)xh)[2 * i + 1] = p1;
    }
    for (int i = tid; i < KC * HID; i += 256) {
        int k = i >> 9;
        int n = i & (HID - 1);
        wt[n * KC + k] = (_Float16)We[i];
    }
#endif
    __syncthreads();

    // ---------------- main loop: H = x_tile @ We, f16 WMMA, f32 accum -----
    v8f acc[4][4];
    #pragma unroll
    for (int m = 0; m < 4; ++m)
        #pragma unroll
        for (int n = 0; n < 4; ++n)
            #pragma unroll
            for (int j = 0; j < 8; ++j) acc[m][n][j] = 0.0f;

    for (int kc = 0; kc < NKC; ++kc) {
        const int cur = kc & 1;
        const int nxt = (kc + 1) & 1;

#ifdef HAVE_TDM
        // kick off DMA of next We chunk while this chunk computes
        if ((kc + 1 < NKC) && wave == 0) {
            tdm_load_2d_f32((unsigned)(uintptr_t)(wstage + nxt * (KC * HID)),
                            We + (size_t)(kc + 1) * KC * HID, HID, KC, HID);
        }
#endif
        const _Float16* wtc = wt + cur * (HID * KC);

        #pragma unroll
        for (int m = 0; m < 4; ++m) {
            // A fragment: lanes 0-15 hold K=0..15, lanes 16-31 K=16..31
            v16h a;
            const int row = m * 16 + l16;
            const _Float16* arow = xh + row * D + kc * KC;
            #pragma unroll
            for (int i = 0; i < 8; ++i) {
                int k0 = 16 * half + 2 * i;
                h2 p = *(const h2*)(arow + k0);
                a[2 * i]     = p.x;
                a[2 * i + 1] = p.y;
            }
            #pragma unroll
            for (int n = 0; n < 4; ++n) {
                // B fragment: lanes 0-15 hold K=0..15, lanes 16-31 K=16..31
                const int col = wave * 64 + n * 16 + l16;
                const _Float16* bcol = wtc + col * KC + 16 * half;
                v16h b;
                #pragma unroll
                for (int i = 0; i < 8; ++i) {
                    h2 p = *(const h2*)(bcol + 2 * i);
                    b[2 * i]     = p.x;
                    b[2 * i + 1] = p.y;
                }
                acc[m][n] = __builtin_amdgcn_wmma_f32_16x16x32_f16(
                    false, a, false, b, (short)0, acc[m][n], false, false);
            }
        }

        if (kc + 1 < NKC) {
#ifdef HAVE_TDM
            if (wave == 0) __builtin_amdgcn_s_wait_tensorcnt(0);
            __syncthreads();   // DMA data visible; prior readers of wt[nxt] done
            const float* src = wstage + nxt * (KC * HID);
            _Float16*    dst = wt + nxt * (HID * KC);
            for (int i = tid; i < KC * HID; i += 256) {
                int k = i >> 9;
                int n = i & (HID - 1);
                dst[n * KC + k] = (_Float16)src[i];
            }
#else
            __syncthreads();
            const float* wg = We + (size_t)(kc + 1) * KC * HID;
            _Float16*    dst = wt + nxt * (HID * KC);
            for (int i = tid; i < KC * HID; i += 256) {
                int k = i >> 9;
                int n = i & (HID - 1);
                dst[n * KC + k] = (_Float16)wg[i];
            }
#endif
            __syncthreads();
        }
    }

    // ---- epilogue: logits[row] = sum_n tanh(h + be[n]) * Wr[n] ----------
    // C layout: element j -> M = m*16 + j + 8*half, N = col
    #pragma unroll
    for (int m = 0; m < 4; ++m) {
        #pragma unroll
        for (int n = 0; n < 4; ++n) {
            const int col = wave * 64 + n * 16 + l16;
            const float bev = be[col];
            const float wrv = Wr[col];
            #pragma unroll
            for (int j = 0; j < 8; ++j) {
                float t = fast_tanh(acc[m][n][j] + bev) * wrv;
                t += __shfl_xor(t, 1);
                t += __shfl_xor(t, 2);
                t += __shfl_xor(t, 4);
                t += __shfl_xor(t, 8);
                if (l16 == 0)
                    atomicAdd(&logits[m * 16 + j + 8 * half], t);
            }
        }
    }
    __syncthreads();

    // ---- phase 2: sigmoid -> softmax per window of 4 rows, mix f32 x ----
    const int dch = tid;
    const float brv = br[0];
    float* outg = out + (size_t)blk * 16 * D;
    for (int s = 0; s < 16; ++s) {
        float sig[4];
        float mx = -1e30f;
        #pragma unroll
        for (int w = 0; w < 4; ++w) {
            float lg = logits[4 * s + w] + brv;
            sig[w] = 1.0f / (1.0f + __expf(-lg));
            mx = fmaxf(mx, sig[w]);
        }
        float ws[4];
        float den = 0.0f;
        #pragma unroll
        for (int w = 0; w < 4; ++w) { ws[w] = __expf(sig[w] - mx); den += ws[w]; }
        const float inv = 1.0f / den;
        float o = 0.0f;
        #pragma unroll
        for (int w = 0; w < 4; ++w)
            o += (ws[w] * inv) * xs[(4 * s + w) * D + dch];
        outg[s * D + dch] = o;
    }
}

extern "C" void kernel_launch(void* const* d_in, const int* in_sizes, int n_in,
                              void* d_out, int out_size, void* d_ws, size_t ws_size,
                              hipStream_t stream) {
    const float* x  = (const float*)d_in[0];
    const float* We = (const float*)d_in[1];
    const float* be = (const float*)d_in[2];
    const float* Wr = (const float*)d_in[3];
    const float* br = (const float*)d_in[4];
    float* out = (float*)d_out;

    const int b = 32, l = 8192;
    const int total_rows = b * l;             // 262144
    const int nblocks = total_rows / ROWS;    // 4096

    (void)in_sizes; (void)n_in; (void)out_size; (void)d_ws; (void)ws_size;

    // gfx1250 WGP has 320KB LDS; we use ~288.25KB per workgroup
    (void)hipFuncSetAttribute((const void*)softpool1d_wmma_kernel,
                              hipFuncAttributeMaxDynamicSharedMemorySize,
                              (int)SM_SIZE);

    softpool1d_wmma_kernel<<<nblocks, 256, SM_SIZE, stream>>>(x, We, be, Wr, br, out);
}
